// VGG16RoIHead_4930622456292
// MI455X (gfx1250) — matrix operand order repaired and checked by reading, not compiled, not run on hardware
//
#include <hip/hip_runtime.h>
#include <hip/hip_bf16.h>
#include <math.h>

// ---------------- problem constants ----------------
#define R_      256
#define C_      512
#define H_      50
#define W_      62
#define PS_     7
#define K6      (C_ * PS_ * PS_)   // 25088
#define FC_     4096
#define NLOC    84                 // N_CLASS * 4
#define NSCORE  21
#define SCALE_  (1.0f / 16.0f)

// ---------------- vector types (trivial, union-safe) ----------------
typedef __attribute__((ext_vector_type(16))) __bf16       v16bf;
typedef __attribute__((ext_vector_type(8)))  float        v8f;
typedef __attribute__((ext_vector_type(4)))  unsigned int u32x4;
typedef __attribute__((ext_vector_type(4)))  float        f32x4;
typedef __attribute__((ext_vector_type(4)))  __bf16       bf16x4;

union Frag16 {                 // one WMMA 16-bit operand (8 VGPRs)
  v16bf          v;
  bf16x4         b4[4];
  u32x4          q[2];
  unsigned short h[16];
};

static __device__ __forceinline__ float bf2f(unsigned u16) {
  union { unsigned u; float f; } c; c.u = u16 << 16;
  return c.f;
}

// ---------------- kernel 1: RoI max pool -> bf16 ----------------
// output layout matches reference: pool[r, c*49 + ph*7 + pw]
__global__ void roi_pool_bf16(const float* __restrict__ x,
                              const float* __restrict__ rois,
                              const int*   __restrict__ roi_idx,
                              __bf16* __restrict__ pool) {
  int idx = blockIdx.x * blockDim.x + threadIdx.x;
  const int total = R_ * C_ * PS_ * PS_;
  if (idx >= total) return;

  int r   = idx / (C_ * PS_ * PS_);
  int rem = idx % (C_ * PS_ * PS_);
  int c   = rem / (PS_ * PS_);
  int p   = rem % (PS_ * PS_);
  int ph  = p / PS_;
  int pw  = p % PS_;

  float y1 = rois[4 * r + 0], x1 = rois[4 * r + 1];
  float y2 = rois[4 * r + 2], x2 = rois[4 * r + 3];
  int rs_h = (int)roundf(y1 * SCALE_);
  int re_h = (int)roundf(y2 * SCALE_);
  int rs_w = (int)roundf(x1 * SCALE_);
  int re_w = (int)roundf(x2 * SCALE_);
  float bsh = (float)max(re_h - rs_h + 1, 1) / (float)PS_;
  float bsw = (float)max(re_w - rs_w + 1, 1) / (float)PS_;

  int h0 = min(max((int)floorf((float)ph * bsh)       + rs_h, 0), H_);
  int h1 = min(max((int)ceilf ((float)(ph + 1) * bsh) + rs_h, 0), H_);
  int w0 = min(max((int)floorf((float)pw * bsw)       + rs_w, 0), W_);
  int w1 = min(max((int)ceilf ((float)(pw + 1) * bsw) + rs_w, 0), W_);

  float v = 0.0f;
  if (h1 > h0 && w1 > w0) {
    int b = roi_idx[r];                      // batch == 1 in practice
    const float* xc = x + ((size_t)b * C_ + c) * (H_ * W_);
    float m = -INFINITY;
    for (int h = h0; h < h1; ++h)
      for (int w = w0; w < w1; ++w)
        m = fmaxf(m, xc[h * W_ + w]);
    v = m;
  }
  pool[idx] = (__bf16)v;
}

// ---------------- kernel 2: bf16 WMMA GEMM, fused bias+ReLU ----------------
// Out[m,n] = relu( sum_k A_bf16[m,k] * W_f32[n,k] + bias[n] )  -> bf16
// A: M x K (bf16, row-major). W: N x K (fp32, row-major, converted in-register
// with packed v_cvt_pk_bf16_f32). One wave computes a 64(M) x 64(N) tile:
// four A fragments share every converted B fragment -> 16 WMMA per K-chunk
// against 32 packed converts. K is a template constant so all row strides fold
// into the 24-bit instruction offsets; the K-loop advances just two pointers.
// Block = 8 waves = 8 tiles. Requires: M % 64 == 0, N % 64 == 0, K % 32 == 0.
template <int K>
__global__ void __launch_bounds__(256)
gemm_bf16_relu(const __bf16* __restrict__ A,
               const float*  __restrict__ Wt,
               const float*  __restrict__ bias,
               __bf16*       __restrict__ Out,
               int M, int N) {
  const int lane   = threadIdx.x & 31;
  const int wave   = threadIdx.x >> 5;
  const int tilesM = M >> 6;                 // 64-row tiles

  int tile = blockIdx.x * 8 + wave;          // wave-uniform
  int tm = tile % tilesM;
  int tn = tile / tilesM;
  const int m0 = tm << 6;
  const int n0 = tn << 6;
  if (n0 >= N) return;                       // wave-uniform guard

  const int half = lane >> 4;                // 0: lanes 0-15, 1: lanes 16-31
  const int lm   = lane & 15;

  // A-operand (16-bit, 16x32): lane holds row; VGPR0-7 = K[16*half .. +15]
  // (contiguous, matches B). Rows m0+16*mi+lm reached via constant offset.
  const __bf16* aPtr = A + (size_t)(m0 + lm) * K + half * 16;

  // B-operand (16-bit, 32x16): lane holds column n0+16*j+lm; half-wave 0 =
  // K 0-15, half-wave 1 = K 16-31 -> one contiguous 16-float read of W row n.
  const float* wPtr = Wt + (size_t)(n0 + lm) * K + half * 16;

  v8f acc[4][4];
#pragma unroll
  for (int mi = 0; mi < 4; ++mi)
#pragma unroll
    for (int j = 0; j < 4; ++j) acc[mi][j] = v8f{};

  for (int k0 = 0; k0 < K; k0 += 32) {
    Frag16 a[4];
#pragma unroll
    for (int mi = 0; mi < 4; ++mi) {         // constant offsets: mi*16*K*2 B
      a[mi].q[0] = *(const u32x4*)(aPtr + mi * 16 * K);
      a[mi].q[1] = *(const u32x4*)(aPtr + mi * 16 * K + 8);
    }

#pragma unroll
    for (int j = 0; j < 4; ++j) {            // constant offsets: j*16*K*4 B
      const float* wp = wPtr + j * 16 * K;
      __builtin_prefetch(wp + 32, 0, 1);     // next K-chunk of weight stream
      f32x4 w0 = *(const f32x4*)(wp + 0);
      f32x4 w1 = *(const f32x4*)(wp + 4);
      f32x4 w2 = *(const f32x4*)(wp + 8);
      f32x4 w3 = *(const f32x4*)(wp + 12);

      Frag16 b;                              // packed f32 -> bf16 (RNE)
      b.b4[0] = __builtin_convertvector(w0, bf16x4);
      b.b4[1] = __builtin_convertvector(w1, bf16x4);
      b.b4[2] = __builtin_convertvector(w2, bf16x4);
      b.b4[3] = __builtin_convertvector(w3, bf16x4);

#pragma unroll
      for (int mi = 0; mi < 4; ++mi)
        acc[mi][j] = __builtin_amdgcn_wmma_f32_16x16x32_bf16(
            false, a[mi].v, false, b.v, (short)0, acc[mi][j], false, false);
    }

    aPtr += 32;
    wPtr += 32;
  }

  // D layout: lanes 0-15 -> N=lm, M = mBase + v; lanes 16-31 -> M = mBase+8+v.
  const int mBase = m0 + half * 8;
#pragma unroll
  for (int j = 0; j < 4; ++j) {
    const int n = n0 + j * 16 + lm;
    const float bv = bias[n];
#pragma unroll
    for (int mi = 0; mi < 4; ++mi) {
#pragma unroll
      for (int v = 0; v < 8; ++v) {
        float val = fmaxf(acc[mi][j][v] + bv, 0.0f);
        Out[(size_t)(mBase + 16 * mi + v) * N + n] = (__bf16)val;
      }
    }
  }
}

// ---------------- kernel 3: small heads (fp32 out) ----------------
__global__ void head_gemv(const unsigned short* __restrict__ A,   // R x FC bf16
                          const float*          __restrict__ Wt,  // Nout x FC
                          const float*          __restrict__ bias,
                          float*                __restrict__ out,
                          int Nout) {
  int idx = blockIdx.x * blockDim.x + threadIdx.x;
  if (idx >= R_ * Nout) return;
  int m = idx / Nout;
  int n = idx % Nout;
  const unsigned short* a = A + (size_t)m * FC_;
  const float* w = Wt + (size_t)n * FC_;
  float s = 0.0f;
  for (int k = 0; k < FC_; k += 8) {
    u32x4 av = *(const u32x4*)(a + k);       // 8 bf16
    f32x4 w0 = *(const f32x4*)(w + k);
    f32x4 w1 = *(const f32x4*)(w + k + 4);
#pragma unroll
    for (int t = 0; t < 4; ++t) {
      s += bf2f(av[t] & 0xFFFFu)                 * w0[t];
      s += bf2f(av[t] >> 16)                     * w1[t];
    }
  }
  out[idx] = s + bias[n];
}

// ---------------- launcher ----------------
extern "C" void kernel_launch(void* const* d_in, const int* in_sizes, int n_in,
                              void* d_out, int out_size, void* d_ws, size_t ws_size,
                              hipStream_t stream) {
  const float* x       = (const float*)d_in[0];
  const float* rois    = (const float*)d_in[1];
  const int*   roi_idx = (const int*)  d_in[2];
  const float* fc6_w   = (const float*)d_in[3];
  const float* fc6_b   = (const float*)d_in[4];
  const float* fc7_w   = (const float*)d_in[5];
  const float* fc7_b   = (const float*)d_in[6];
  const float* cls_w   = (const float*)d_in[7];
  const float* cls_b   = (const float*)d_in[8];
  const float* sc_w    = (const float*)d_in[9];
  const float* sc_b    = (const float*)d_in[10];

  // workspace: pool (R x 25088 bf16) | fc6 (R x 4096 bf16) | fc7 (R x 4096 bf16)
  __bf16* pool_bf = (__bf16*)d_ws;
  __bf16* fc6_bf  = pool_bf + (size_t)R_ * K6;
  __bf16* fc7_bf  = fc6_bf + (size_t)R_ * FC_;
  float* out = (float*)d_out;

  {
    int total = R_ * C_ * PS_ * PS_;
    roi_pool_bf16<<<(total + 255) / 256, 256, 0, stream>>>(x, rois, roi_idx, pool_bf);
  }
  {
    int tiles = (R_ / 64) * (FC_ / 64);   // 256 wave-tiles, 8 per block
    gemm_bf16_relu<K6><<<tiles / 8, 256, 0, stream>>>(pool_bf, fc6_w, fc6_b,
                                                      fc6_bf, R_, FC_);
  }
  {
    int tiles = (R_ / 64) * (FC_ / 64);
    gemm_bf16_relu<FC_><<<tiles / 8, 256, 0, stream>>>(fc6_bf, fc7_w, fc7_b,
                                                       fc7_bf, R_, FC_);
  }
  {
    int total = R_ * NLOC;
    head_gemv<<<(total + 255) / 256, 256, 0, stream>>>(
        (const unsigned short*)fc7_bf, cls_w, cls_b, out, NLOC);
  }
  {
    int total = R_ * NSCORE;
    head_gemv<<<(total + 255) / 256, 256, 0, stream>>>(
        (const unsigned short*)fc7_bf, sc_w, sc_b, out + (size_t)R_ * NLOC, NSCORE);
  }
}
